// AttentionRelPos_31078383354076
// MI455X (gfx1250) — matrix operand order repaired and checked by reading, not compiled, not run on hardware
//
#include <hip/hip_runtime.h>

// ---------------------------------------------------------------------------
// Problem constants (from reference): B=2, T=8, H=W=14, DIM=768, HEADS=12
// ---------------------------------------------------------------------------
constexpr int CB      = 2;
constexpr int CT      = 8;
constexpr int CH      = 14;
constexpr int CW      = 14;
constexpr int CS      = CH * CW;          // 196
constexpr int CN      = CT * CS;          // 1568 tokens
constexpr int CDIM    = 768;
constexpr int CHEADS  = 12;
constexpr int CHD     = 64;
constexpr int CBH     = CB * CHEADS;      // 24
constexpr int CBN     = CB * CN;          // 3136 rows
constexpr float CSCALE = 0.125f;          // 64^-0.5

typedef __attribute__((ext_vector_type(16))) __bf16 v16bf;
typedef __attribute__((ext_vector_type(8)))  float  v8f;
typedef __attribute__((ext_vector_type(4)))  int    v4i;

#define DEV static __device__ __forceinline__

// ---------------------------------------------------------------------------
// CDNA5 async global->LDS path (guarded; falls back to plain copies)
// Probe-discovered signature: (int4 AS1*, int4 AS3*, imm offset, imm cpol)
// ---------------------------------------------------------------------------
#if defined(__has_builtin)
#  if __has_builtin(__builtin_amdgcn_global_load_async_to_lds_b128) && \
      __has_builtin(__builtin_amdgcn_s_wait_asynccnt)
#    define HAVE_ASYNC 1
#  endif
#endif
#ifndef HAVE_ASYNC
#  define HAVE_ASYNC 0
#endif

#if HAVE_ASYNC
typedef __attribute__((address_space(1))) v4i* gv4i_p;
typedef __attribute__((address_space(3))) v4i* lv4i_p;
#endif

DEV void async_copy16(const __bf16* g, __bf16* l) {
#if HAVE_ASYNC
  __builtin_amdgcn_global_load_async_to_lds_b128((gv4i_p)g, (lv4i_p)l, 0, 0);
#else
  *reinterpret_cast<uint4*>(l) = *reinterpret_cast<const uint4*>(g);
#endif
}

DEV void async_wait() {
#if HAVE_ASYNC
  __builtin_amdgcn_s_wait_asynccnt(0);
#endif
}

// ---------------------------------------------------------------------------
// WMMA wrapper: D = A(16x32 bf16) * B(32x16 bf16) + C(16x16 f32)
// ---------------------------------------------------------------------------
DEV v8f wmma_bf16(v16bf a, v16bf b, v8f c) {
  return __builtin_amdgcn_wmma_f32_16x16x32_bf16(
      /*neg_a=*/false, a, /*neg_b=*/false, b,
      /*c_mod=*/(short)0, c, /*reuse_a=*/false, /*reuse_b=*/false);
}

DEV v8f zero8() {
  v8f z;
#pragma unroll
  for (int i = 0; i < 8; ++i) z[i] = 0.0f;
  return z;
}

// ---------------------------------------------------------------------------
// Fragment loaders (this file's contiguous k order; A and B use the same).
// A 16x32 bf16: lane = half*16 + m; e=0..15 -> K=half*16+e (16 contiguous elems)
// B 32x16 bf16: lane = half*16 + n; e=0..15 -> K=half*16+e (16 contiguous elems)
// C/D 16x16 f32: element r -> row M = r + 8*half, col N = lane&15
// ---------------------------------------------------------------------------
DEV v16bf load_a_bf16(const __bf16* __restrict__ row, int half) {
  union U { uint4 u; __bf16 h[8]; } lo, hi;
  lo.u = *reinterpret_cast<const uint4*>(row + half * 16);
  hi.u = *reinterpret_cast<const uint4*>(row + half * 16 + 8);
  v16bf a;
#pragma unroll
  for (int e = 0; e < 8; ++e) { a[e] = lo.h[e]; a[e + 8] = hi.h[e]; }
  return a;
}

DEV v16bf load_b_bf16(const __bf16* __restrict__ p) {
  union U { uint4 u; __bf16 h[8]; } lo, hi;
  lo.u = *reinterpret_cast<const uint4*>(p);
  hi.u = *reinterpret_cast<const uint4*>(p + 8);
  v16bf b;
#pragma unroll
  for (int e = 0; e < 8; ++e) { b[e] = lo.h[e]; b[e + 8] = hi.h[e]; }
  return b;
}

// ---------------------------------------------------------------------------
// Kernel 0: one-time fp32 -> bf16 conversion (8 elems / thread)
// ---------------------------------------------------------------------------
__global__ __launch_bounds__(256) void cvt_bf16_kernel(
    const float* __restrict__ src, __bf16* __restrict__ dst, int n8) {
  int i = blockIdx.x * blockDim.x + threadIdx.x;
  if (i >= n8) return;
  const float4* s = reinterpret_cast<const float4*>(src) + (size_t)i * 2;
  float4 a = s[0], b = s[1];
  union U { uint4 u; __bf16 h[8]; } o;
  o.h[0] = (__bf16)a.x; o.h[1] = (__bf16)a.y;
  o.h[2] = (__bf16)a.z; o.h[3] = (__bf16)a.w;
  o.h[4] = (__bf16)b.x; o.h[5] = (__bf16)b.y;
  o.h[6] = (__bf16)b.z; o.h[7] = (__bf16)b.w;
  reinterpret_cast<uint4*>(dst)[i] = o.u;
}

// ---------------------------------------------------------------------------
// Kernel 1: QKV GEMM  qkv[3136,2304] = Xb * Wb^T (both bf16), scattered into
//   Q bf16 [bh][n][64], K bf16 [bh][n][64], Vt bf16 [bh][64][n]
// One wave per 16x32 output tile (A fragment reused by 2 WMMAs).
// ---------------------------------------------------------------------------
__global__ __launch_bounds__(256) void qkv_gemm_kernel(
    const __bf16* __restrict__ xb, const __bf16* __restrict__ wb,
    __bf16* __restrict__ qd, __bf16* __restrict__ kd,
    __bf16* __restrict__ vtd) {
  const int TN = (3 * CDIM) / 32;  // 72
  int wave = blockIdx.x * (blockDim.x >> 5) + (threadIdx.x >> 5);
  if (wave >= (CBN / 16) * TN) return;
  const int tm = wave / TN, tn = wave % TN;
  const int lane = threadIdx.x & 31, half = lane >> 4, ml = lane & 15;

  const __bf16* arow  = xb + (size_t)(tm * 16 + ml) * CDIM;
  const __bf16* wrow0 = wb + (size_t)(tn * 32 + ml) * CDIM;       // cols +0..15
  const __bf16* wrow1 = wb + (size_t)(tn * 32 + 16 + ml) * CDIM;  // cols +16..31
  v8f acc0 = zero8(), acc1 = zero8();
#pragma unroll 4
  for (int k0 = 0; k0 < CDIM; k0 += 32) {
    if (k0 + 32 < CDIM) __builtin_prefetch(arow + k0 + 64, 0, 1);
    v16bf a  = load_a_bf16(arow + k0, half);
    v16bf b0 = load_b_bf16(wrow0 + k0 + half * 16);
    v16bf b1 = load_b_bf16(wrow1 + k0 + half * 16);
    acc0 = wmma_bf16(a, b0, acc0);
    acc1 = wmma_bf16(a, b1, acc1);
  }

  // Epilogue scatter. 32-wide tile never straddles a head (64) or sel (768).
  const int col0 = tn * 32;
  const int sel  = col0 / CDIM;
  const int rem0 = col0 % CDIM;
  const int head = rem0 / CHD;
  const int dA   = (rem0 % CHD) + ml;
  const int dB   = dA + 16;
#pragma unroll
  for (int r = 0; r < 8; ++r) {
    const int m = tm * 16 + r + 8 * half;
    const int b_ = m / CN, n_ = m % CN;
    const int bh = b_ * CHEADS + head;
    const __bf16 vA = (__bf16)acc0[r];
    const __bf16 vB = (__bf16)acc1[r];
    if (sel == 0) {
      qd[((size_t)bh * CN + n_) * CHD + dA] = vA;
      qd[((size_t)bh * CN + n_) * CHD + dB] = vB;
    } else if (sel == 1) {
      kd[((size_t)bh * CN + n_) * CHD + dA] = vA;
      kd[((size_t)bh * CN + n_) * CHD + dB] = vB;
    } else {
      vtd[((size_t)bh * CHD + dA) * CN + n_] = vA;
      vtd[((size_t)bh * CHD + dB) * CN + n_] = vB;
    }
  }
}

// ---------------------------------------------------------------------------
// Kernel 2: decomposed rel-pos bias tables (unscaled q, as in reference)
//   rel_h[bh][n][14], rel_w[bh][n][14], rel_t[bh][n][8]
// ---------------------------------------------------------------------------
__global__ __launch_bounds__(256) void relbias_kernel(
    const __bf16* __restrict__ q, const float* __restrict__ rph,
    const float* __restrict__ rpw, const float* __restrict__ rpt,
    float* __restrict__ relh, float* __restrict__ relw,
    float* __restrict__ relt) {
  const int TOT = CBH * CN * (CH + CW + CT);  // 36 outputs per (bh,n)
  int idx = blockIdx.x * blockDim.x + threadIdx.x;
  if (idx >= TOT) return;
  const int o  = idx % 36;
  const int qi = idx / 36;               // bh*N + n
  const int n  = qi % CN;
  const int tq = n / CS, rem = n % CS, hq = rem / CW, wq = rem % CW;

  const float* tab; float* dst; int k, width, row;
  if (o < CH)           { k = o;      row = hq - k + (CH - 1); tab = rph; dst = relh; width = CH; }
  else if (o < CH + CW) { k = o - CH; row = wq - k + (CW - 1); tab = rpw; dst = relw; width = CW; }
  else                  { k = o - 28; row = tq - k + (CT - 1); tab = rpt; dst = relt; width = CT; }

  const __bf16* qv = q + (size_t)qi * CHD;
  const float* rv = tab + (size_t)row * CHD;
  float s = 0.0f;
#pragma unroll
  for (int d0 = 0; d0 < CHD; d0 += 8) {
    union U { uint4 u; __bf16 h[8]; } qq;
    qq.u = *reinterpret_cast<const uint4*>(qv + d0);
    float4 r0 = *reinterpret_cast<const float4*>(rv + d0);
    float4 r1 = *reinterpret_cast<const float4*>(rv + d0 + 4);
    s += (float)qq.h[0] * r0.x + (float)qq.h[1] * r0.y +
         (float)qq.h[2] * r0.z + (float)qq.h[3] * r0.w +
         (float)qq.h[4] * r1.x + (float)qq.h[5] * r1.y +
         (float)qq.h[6] * r1.z + (float)qq.h[7] * r1.w;
  }
  dst[(size_t)qi * width + k] = s;
}

// ---------------------------------------------------------------------------
// Kernel 3: flash attention. 2 waves / block = two q-tiles of one (b,head)
// sharing async-staged K/V blocks in LDS.
// S = scale*QK^T + bias; online softmax over 32-key blocks; O = P*V.
// Output: aout bf16 [b][n][head*64+d]  (== transpose(0,2,1,3) layout)
// ---------------------------------------------------------------------------
__global__ __launch_bounds__(64) void attn_kernel(
    const __bf16* __restrict__ qm, const __bf16* __restrict__ km,
    const __bf16* __restrict__ vtm, const float* __restrict__ relh,
    const float* __restrict__ relw, const float* __restrict__ relt,
    __bf16* __restrict__ aout) {
  __shared__ __bf16 Kbuf[32 * 64];      // [key_local][d]      4 KB
  __shared__ __bf16 Vbuf[64 * 32];      // [d][key_local]      4 KB
  __shared__ __bf16 pls[2][16 * 32];    // per-wave P staging  2 KB

  const int tid = threadIdx.x;
  const int widx = tid >> 5, lane = tid & 31, half = lane >> 4, nl = lane & 15;
  const int blk = blockIdx.x;            // bh * 49 + bq
  const int bq = blk % (CN / 32);
  const int bh = blk / (CN / 32);
  const int b_ = bh / CHEADS, head = bh % CHEADS;
  const int qt = bq * 2 + widx;
  const int q0 = qt * 16;

  const __bf16* qb = qm + (size_t)bh * CN * CHD;
  const __bf16* kb = km + (size_t)bh * CN * CHD;
  const __bf16* vb = vtm + (size_t)bh * CHD * CN;
  const float* rhb = relh + (size_t)bh * CN * CH;
  const float* rwb = relw + (size_t)bh * CN * CW;
  const float* rtb = relt + (size_t)bh * CN * CT;

  // Preload Q fragments for this wave's tile: d-blocks [0,32) and [32,64)
  const __bf16* qrow = qb + (size_t)(q0 + nl) * CHD;
  v16bf qa0 = load_a_bf16(qrow, half);
  v16bf qa1 = load_a_bf16(qrow + 32, half);

  v8f o0 = zero8(), o1 = zero8(), o2 = zero8(), o3 = zero8();
  float mrow[8], lrow[8];
#pragma unroll
  for (int r = 0; r < 8; ++r) { mrow[r] = -3.0e38f; lrow[r] = 0.0f; }

  for (int j0 = 0; j0 < CN; j0 += 32) {
    // ---- cooperative async stage: K block (4KB contiguous) + V block (4KB)
    const __bf16* gk = kb + (size_t)j0 * CHD;
    if (j0 + 32 < CN) __builtin_prefetch(gk + 32 * CHD, 0, 1);
#pragma unroll
    for (int i = 0; i < 4; ++i) {
      const int c = i * 64 + tid;        // 16B chunk id, 0..255
      async_copy16(gk + c * 8, &Kbuf[c * 8]);
    }
#pragma unroll
    for (int i = 0; i < 4; ++i) {
      const int c = i * 64 + tid;        // 0..255
      const int row = c >> 2, ch = c & 3;
      async_copy16(vb + (size_t)row * CN + j0 + ch * 8,
                   &Vbuf[row * 32 + ch * 8]);
    }
    async_wait();
    __syncthreads();

    // ---- S tiles: keys [j0,j0+16) and [j0+16,j0+32), fragments from LDS
    v8f s0 = zero8(), s1 = zero8();
    {
      v16bf bk;
      bk = load_b_bf16(&Kbuf[nl * 64 + half * 16]);        s0 = wmma_bf16(qa0, bk, s0);
      bk = load_b_bf16(&Kbuf[nl * 64 + 32 + half * 16]);   s0 = wmma_bf16(qa1, bk, s0);
      bk = load_b_bf16(&Kbuf[(16 + nl) * 64 + half * 16]); s1 = wmma_bf16(qa0, bk, s1);
      bk = load_b_bf16(&Kbuf[(16 + nl) * 64 + 32 + half * 16]);
      s1 = wmma_bf16(qa1, bk, s1);
    }
    // ---- scale + decomposed rel-pos bias (key coords uniform per lane/frag)
    const int key0 = j0 + nl, key1 = j0 + 16 + nl;
    const int t0 = key0 / CS, e0 = key0 % CS, h0 = e0 / CW, w0 = e0 % CW;
    const int t1 = key1 / CS, e1 = key1 % CS, h1 = e1 / CW, w1 = e1 % CW;
#pragma unroll
    for (int r = 0; r < 8; ++r) {
      const int qq = q0 + r + 8 * half;
      const float bias0 = rhb[qq * CH + h0] + rwb[qq * CW + w0] + rtb[qq * CT + t0];
      const float bias1 = rhb[qq * CH + h1] + rwb[qq * CW + w1] + rtb[qq * CT + t1];
      s0[r] = s0[r] * CSCALE + bias0;
      s1[r] = s1[r] * CSCALE + bias1;
    }
    // ---- online softmax (row r+8*half lives on a 16-lane half)
#pragma unroll
    for (int r = 0; r < 8; ++r) {
      float mx = fmaxf(s0[r], s1[r]);
      mx = fmaxf(mx, __shfl_xor(mx, 1, 32));
      mx = fmaxf(mx, __shfl_xor(mx, 2, 32));
      mx = fmaxf(mx, __shfl_xor(mx, 4, 32));
      mx = fmaxf(mx, __shfl_xor(mx, 8, 32));
      const float mnew = fmaxf(mrow[r], mx);
      const float corr = __expf(mrow[r] - mnew);
      const float p0 = __expf(s0[r] - mnew);
      const float p1 = __expf(s1[r] - mnew);
      float ps = p0 + p1;
      ps += __shfl_xor(ps, 1, 32);
      ps += __shfl_xor(ps, 2, 32);
      ps += __shfl_xor(ps, 4, 32);
      ps += __shfl_xor(ps, 8, 32);
      lrow[r] = lrow[r] * corr + ps;
      mrow[r] = mnew;
      o0[r] *= corr; o1[r] *= corr; o2[r] *= corr; o3[r] *= corr;
      const int row = r + 8 * half;
      pls[widx][row * 32 + nl] = (__bf16)p0;
      pls[widx][row * 32 + 16 + nl] = (__bf16)p1;
    }
    // ---- P * V  (pls is wave-local; same-wave LDS ops are in-order)
    {
      v16bf pa = load_a_bf16(&pls[widx][nl * 32], half);
      v16bf bv;
      bv = load_b_bf16(&Vbuf[(0 + nl) * 32 + half * 16]);  o0 = wmma_bf16(pa, bv, o0);
      bv = load_b_bf16(&Vbuf[(16 + nl) * 32 + half * 16]); o1 = wmma_bf16(pa, bv, o1);
      bv = load_b_bf16(&Vbuf[(32 + nl) * 32 + half * 16]); o2 = wmma_bf16(pa, bv, o2);
      bv = load_b_bf16(&Vbuf[(48 + nl) * 32 + half * 16]); o3 = wmma_bf16(pa, bv, o3);
    }
    __syncthreads();  // all waves done reading Kbuf/Vbuf before restaging
  }

  // ---- epilogue: normalize and store bf16 into [b][n][head*64+d]
#pragma unroll
  for (int r = 0; r < 8; ++r) {
    const float inv = 1.0f / lrow[r];
    const int qq = q0 + r + 8 * half;
    __bf16* dst = aout + ((size_t)b_ * CN + qq) * CDIM + head * CHD;
    dst[0 + nl]  = (__bf16)(o0[r] * inv);
    dst[16 + nl] = (__bf16)(o1[r] * inv);
    dst[32 + nl] = (__bf16)(o2[r] * inv);
    dst[48 + nl] = (__bf16)(o3[r] * inv);
  }
}

// ---------------------------------------------------------------------------
// Kernel 4: out[3136,768] = aout_bf16 * proj_wb^T + proj_b  (fp32 out)
// One wave per 16x32 output tile.
// ---------------------------------------------------------------------------
__global__ __launch_bounds__(256) void proj_gemm_kernel(
    const __bf16* __restrict__ a, const __bf16* __restrict__ wb,
    const float* __restrict__ bias, float* __restrict__ out) {
  const int TN = CDIM / 32;  // 24
  int wave = blockIdx.x * (blockDim.x >> 5) + (threadIdx.x >> 5);
  if (wave >= (CBN / 16) * TN) return;
  const int tm = wave / TN, tn = wave % TN;
  const int lane = threadIdx.x & 31, half = lane >> 4, ml = lane & 15;

  const __bf16* arow  = a + (size_t)(tm * 16 + ml) * CDIM;
  const __bf16* wrow0 = wb + (size_t)(tn * 32 + ml) * CDIM;
  const __bf16* wrow1 = wb + (size_t)(tn * 32 + 16 + ml) * CDIM;
  v8f acc0 = zero8(), acc1 = zero8();
#pragma unroll 4
  for (int k0 = 0; k0 < CDIM; k0 += 32) {
    if (k0 + 32 < CDIM) __builtin_prefetch(arow + k0 + 64, 0, 1);
    v16bf af = load_a_bf16(arow + k0, half);
    v16bf b0 = load_b_bf16(wrow0 + k0 + half * 16);
    v16bf b1 = load_b_bf16(wrow1 + k0 + half * 16);
    acc0 = wmma_bf16(af, b0, acc0);
    acc1 = wmma_bf16(af, b1, acc1);
  }
  const int colA = tn * 32 + ml;
  const int colB = colA + 16;
  const float bbA = bias[colA], bbB = bias[colB];
#pragma unroll
  for (int r = 0; r < 8; ++r) {
    const int m = tm * 16 + r + 8 * half;
    out[(size_t)m * CDIM + colA] = acc0[r] + bbA;
    out[(size_t)m * CDIM + colB] = acc1[r] + bbB;
  }
}

// ---------------------------------------------------------------------------
// Launch. Inputs: x, qkv_w, proj_w, proj_b, rel_pos_h, rel_pos_w, rel_pos_t
// ---------------------------------------------------------------------------
extern "C" void kernel_launch(void* const* d_in, const int* in_sizes, int n_in,
                              void* d_out, int out_size, void* d_ws,
                              size_t ws_size, hipStream_t stream) {
  const float* x      = (const float*)d_in[0];
  const float* qkv_w  = (const float*)d_in[1];
  const float* proj_w = (const float*)d_in[2];
  const float* proj_b = (const float*)d_in[3];
  const float* rph    = (const float*)d_in[4];
  const float* rpw    = (const float*)d_in[5];
  const float* rpt    = (const float*)d_in[6];
  float* out = (float*)d_out;

  // Workspace carve-up (bytes)
  char* ws = (char*)d_ws;
  constexpr size_t xbSz   = (size_t)CBN * CDIM * 2;            // 4,816,896
  constexpr size_t wqSz   = (size_t)3 * CDIM * CDIM * 2;       // 3,538,944
  constexpr size_t wpSz   = (size_t)CDIM * CDIM * 2;           // 1,179,648
  constexpr size_t qkvSz  = (size_t)CBH * CN * CHD * 2;        // 4,816,896 each
  constexpr size_t relhSz = (size_t)CBH * CN * CH * 4;
  constexpr size_t relwSz = (size_t)CBH * CN * CW * 4;
  constexpr size_t reltSz = (size_t)CBH * CN * CT * 4;

  size_t off = 0;
  __bf16* xb    = (__bf16*)(ws + off); off += xbSz;
  __bf16* wqkvb = (__bf16*)(ws + off); off += wqSz;
  __bf16* wprojb= (__bf16*)(ws + off); off += wpSz;
  __bf16* qd    = (__bf16*)(ws + off); off += qkvSz;
  __bf16* kd    = (__bf16*)(ws + off); off += qkvSz;
  __bf16* vtd   = (__bf16*)(ws + off); off += qkvSz;
  float*  relh  = (float*)(ws + off);  off += relhSz;
  float*  relw  = (float*)(ws + off);  off += relwSz;
  float*  relt  = (float*)(ws + off);  off += reltSz;
  __bf16* aoutp = (__bf16*)(ws + off);

  // 0) one-time bf16 conversions
  {
    const int n8x = CBN * CDIM / 8;           // 301056
    const int n8q = 3 * CDIM * CDIM / 8;      // 221184
    const int n8p = CDIM * CDIM / 8;          // 73728
    cvt_bf16_kernel<<<n8x / 256, 256, 0, stream>>>(x, xb, n8x);
    cvt_bf16_kernel<<<n8q / 256, 256, 0, stream>>>(qkv_w, wqkvb, n8q);
    cvt_bf16_kernel<<<n8p / 256, 256, 0, stream>>>(proj_w, wprojb, n8p);
  }
  // 1) QKV GEMM: (3136/16)*(2304/32) = 14112 waves, 8 waves/block
  {
    const int waves = (CBN / 16) * ((3 * CDIM) / 32);
    qkv_gemm_kernel<<<waves / 8, 256, 0, stream>>>(xb, wqkvb, qd, kd, vtd);
  }
  // 2) rel-pos bias tables
  {
    const int tot = CBH * CN * 36;
    relbias_kernel<<<(tot + 255) / 256, 256, 0, stream>>>(qd, rph, rpw, rpt,
                                                          relh, relw, relt);
  }
  // 3) flash attention: 24 heads*batch x 49 blocks, 2 waves each
  {
    attn_kernel<<<CBH * (CN / 32), 64, 0, stream>>>(qd, kd, vtd, relh, relw,
                                                    relt, aoutp);
  }
  // 4) projection GEMM: (3136/16)*(768/32) = 4704 waves
  {
    const int waves = (CBN / 16) * (CDIM / 32);
    proj_gemm_kernel<<<waves / 8, 256, 0, stream>>>(aoutp, wprojb, proj_b, out);
  }
}
